// QuantizedAlexNetSmall_70583492543062
// MI455X (gfx1250) — compile-verified, matrix operand order checked by reading, not run on hardware
//
#include <hip/hip_runtime.h>
#include <stdint.h>

typedef int v8i __attribute__((ext_vector_type(8)));
typedef signed char s8;

// ---------------------------------------------------------------------------
// Quantize fp32 -> int8 (round-to-nearest-even, clamp [-128,127])
// ---------------------------------------------------------------------------
__global__ void quantize_kernel(const float* __restrict__ x, s8* __restrict__ out, int n) {
  int i = blockIdx.x * 256 + threadIdx.x;
  if (i >= n) return;
  float v = rintf(x[i] * (1.0f / 0.05f));
  v = fminf(fmaxf(v, -128.0f), 127.0f);
  out[i] = (s8)(int)v;
}

// ---------------------------------------------------------------------------
// Pack a row-major [R x K] int8-valued fp32 matrix into CDNA5 WMMA 16x64
// 8-bit fragment blobs. Blob layout: tile (rt,kt) -> 1024 bytes laid out as
// [lane 0..31][8 dwords], element map per ISA 7.12.2 (8-bit A 16x64):
//   row   = rt*16 + (lane & 15)
//   k     = kt*64 + (v>>1)*16 + (lane>>4)*8 + (v&1)*4 + byte
// The same map serves the B operand of the FC GEMMs (rows = out-channels).
// ---------------------------------------------------------------------------
__global__ void pack_frag_kernel(const float* __restrict__ w, s8* __restrict__ out,
                                 int R, int K) {
  int RT = (R + 15) >> 4;
  int KT = (K + 63) >> 6;
  long long total = (long long)RT * KT * 1024;
  long long i = (long long)blockIdx.x * 256 + threadIdx.x;
  if (i >= total) return;
  int b    = (int)(i & 3);
  int v    = (int)((i >> 2) & 7);
  int lane = (int)((i >> 5) & 31);
  long long tile = i >> 10;
  int kt = (int)(tile % KT);
  int rt = (int)(tile / KT);
  int row   = rt * 16 + (lane & 15);
  int khalf = lane >> 4;
  int k = kt * 64 + ((v >> 1) << 4) + (khalf << 3) + ((v & 1) << 2) + b;
  s8 val = 0;
  if (row < R && k < K) val = (s8)(int)w[(long long)row * K + k];
  out[i] = val;
}

// ---------------------------------------------------------------------------
// Implicit-GEMM int8 conv (stride 1). Block = 128 threads = 4 waves, all
// sharing one 16-pixel tile. Each wave owns 4 OC tiles (256 OCs/block).
// Per K-step the im2col B fragment is staged once in LDS, read once per wave,
// and reused by 4 back-to-back WMMAs against streamed weight fragments.
// Weight-tile indices are clamped (not branched) so EXEC stays uniform and
// the hot loop is pure loads + v_wmma.
// ---------------------------------------------------------------------------
template <int KH, int KW, int PAD>
__global__ void conv_wmma_kernel(const s8* __restrict__ act, const s8* __restrict__ wfrag,
                                 const float* __restrict__ bias, s8* __restrict__ out,
                                 int B, int IC, int H, int W, int OC, float M) {
  const int K    = IC * KH * KW;
  const int KT   = (K + 63) >> 6;
  const int NPIX = B * H * W;
  const int OCT  = (OC + 15) >> 4;

  const int wave  = threadIdx.x >> 5;
  const int lane  = threadIdx.x & 31;
  const int pt    = blockIdx.x;                      // pixel tile
  const int oct0  = blockIdx.y * 16 + wave * 4;      // first of 4 oc tiles
  const int khalf = lane >> 4;

  __shared__ int lds_b[256];  // one 16x64 int8 B fragment (1 KB)

  v8i acc0 = {0,0,0,0,0,0,0,0}, acc1 = {0,0,0,0,0,0,0,0};
  v8i acc2 = {0,0,0,0,0,0,0,0}, acc3 = {0,0,0,0,0,0,0,0};

  // clamped weight-tile bases: always-valid loads, no exec divergence
  const int c0 = (oct0 + 0 < OCT) ? oct0 + 0 : OCT - 1;
  const int c1 = (oct0 + 1 < OCT) ? oct0 + 1 : OCT - 1;
  const int c2 = (oct0 + 2 < OCT) ? oct0 + 2 : OCT - 1;
  const int c3 = (oct0 + 3 < OCT) ? oct0 + 3 : OCT - 1;
  const int* ap0 = (const int*)(wfrag + ((long long)c0 * KT << 10)) + lane * 8;
  const int* ap1 = (const int*)(wfrag + ((long long)c1 * KT << 10)) + lane * 8;
  const int* ap2 = (const int*)(wfrag + ((long long)c2 * KT << 10)) + lane * 8;
  const int* ap3 = (const int*)(wfrag + ((long long)c3 * KT << 10)) + lane * 8;

  for (int kt = 0; kt < KT; ++kt) {
    __syncthreads();  // previous iteration done reading lds_b
    {
      // cooperative im2col gather: 128 threads x 8 bytes = 1024 bytes
      const int base = threadIdx.x << 3;
      int lo = 0, hi = 0;
#pragma unroll
      for (int j = 0; j < 8; ++j) {
        const int ii  = base + j;
        const int bb  = ii & 3;
        const int vv  = (ii >> 2) & 7;
        const int ln  = ii >> 5;
        const int kh2 = ln >> 4;
        const int k = kt * 64 + ((vv >> 1) << 4) + (kh2 << 3) + ((vv & 1) << 2) + bb;
        const int p = pt * 16 + (ln & 15);
        int val = 0;
        if (k < K && p < NPIX) {
          const int ic  = k / (KH * KW);          // constant divisor
          const int rem = k - ic * (KH * KW);
          const int kh  = rem / KW;
          const int kw  = rem - kh * KW;
          const int bi  = p / (H * W);
          const int pr  = p - bi * (H * W);
          const int oy  = pr / W;
          const int ox  = pr - oy * W;
          const int iy  = oy - PAD + kh;
          const int ix  = ox - PAD + kw;
          if (iy >= 0 && iy < H && ix >= 0 && ix < W)
            val = act[(((long long)bi * IC + ic) * H + iy) * W + ix];
        }
        if (j < 4) lo |= (val & 0xff) << (8 * j);
        else       hi |= (val & 0xff) << (8 * (j - 4));
      }
      lds_b[base >> 2]       = lo;
      lds_b[(base >> 2) + 1] = hi;
    }
    __syncthreads();

    v8i bfrag;
    {
      const int* bp = lds_b + lane * 8;
#pragma unroll
      for (int j = 0; j < 8; ++j) bfrag[j] = bp[j];
    }
    const long long ko = (long long)kt << 8;  // kt*1024 bytes = kt*256 ints
    v8i a0, a1, a2, a3;
#pragma unroll
    for (int j = 0; j < 8; ++j) a0[j] = ap0[ko + j];
#pragma unroll
    for (int j = 0; j < 8; ++j) a1[j] = ap1[ko + j];
#pragma unroll
    for (int j = 0; j < 8; ++j) a2[j] = ap2[ko + j];
#pragma unroll
    for (int j = 0; j < 8; ++j) a3[j] = ap3[ko + j];
    __builtin_prefetch(ap0 + ko + 256, 0, 0);  // next K-step weight stream
    acc0 = __builtin_amdgcn_wmma_i32_16x16x64_iu8(true, a0, true, bfrag, acc0, false, false);
    acc1 = __builtin_amdgcn_wmma_i32_16x16x64_iu8(true, a1, true, bfrag, acc1, false, false);
    acc2 = __builtin_amdgcn_wmma_i32_16x16x64_iu8(true, a2, true, bfrag, acc2, false, false);
    acc3 = __builtin_amdgcn_wmma_i32_16x16x64_iu8(true, a3, true, bfrag, acc3, false, false);
  }

  // D layout: VGPR r holds (M = r + 8*khalf, N = lane&15)
  const int p = pt * 16 + (lane & 15);
  const int bi = p / (H * W);
  const int pr = p - bi * (H * W);
  const int oy = pr / W;
  const int ox = pr - oy * W;
#pragma unroll
  for (int t = 0; t < 4; ++t) {
    const v8i acc = (t == 0) ? acc0 : (t == 1) ? acc1 : (t == 2) ? acc2 : acc3;
    const int oct = oct0 + t;
    if (oct >= OCT) continue;
#pragma unroll
    for (int r = 0; r < 8; ++r) {
      const int oc = oct * 16 + r + (khalf << 3);
      if (oc < OC && p < NPIX) {
        const int a = acc[r] + (int)bias[oc];
        float o = floorf((float)a * M);
        o = fminf(fmaxf(o, 0.0f), 127.0f);  // requant + ReLU clamp
        out[(((long long)bi * OC + oc) * H + oy) * W + ox] = (s8)(int)o;
      }
    }
  }
}

// ---------------------------------------------------------------------------
// FC GEMM: D[b][oc] = act[b][:] . wf[oc][:]. Each wave owns 4 OC tiles and
// reuses its A fragment (activations) across 4 WMMAs per K-step.
// ---------------------------------------------------------------------------
__global__ void fc_wmma_kernel(const s8* __restrict__ act, const s8* __restrict__ wfrag,
                               const float* __restrict__ bias, s8* __restrict__ out_s8,
                               float* __restrict__ out_f32, int K, int OC, float M,
                               int relu) {
  const int KT  = K >> 6;
  const int OCT = (OC + 15) >> 4;
  const int wave  = threadIdx.x >> 5;
  const int lane  = threadIdx.x & 31;
  const int bt    = blockIdx.x;
  const int oct0  = (blockIdx.y * 4 + wave) * 4;
  const int khalf = lane >> 4;
  const int row   = bt * 16 + (lane & 15);  // batch row for A operand

  const int c0 = (oct0 + 0 < OCT) ? oct0 + 0 : OCT - 1;
  const int c1 = (oct0 + 1 < OCT) ? oct0 + 1 : OCT - 1;
  const int c2 = (oct0 + 2 < OCT) ? oct0 + 2 : OCT - 1;
  const int c3 = (oct0 + 3 < OCT) ? oct0 + 3 : OCT - 1;
  const int* bp0 = (const int*)(wfrag + ((long long)c0 * KT << 10)) + lane * 8;
  const int* bp1 = (const int*)(wfrag + ((long long)c1 * KT << 10)) + lane * 8;
  const int* bp2 = (const int*)(wfrag + ((long long)c2 * KT << 10)) + lane * 8;
  const int* bp3 = (const int*)(wfrag + ((long long)c3 * KT << 10)) + lane * 8;

  v8i acc0 = {0,0,0,0,0,0,0,0}, acc1 = {0,0,0,0,0,0,0,0};
  v8i acc2 = {0,0,0,0,0,0,0,0}, acc3 = {0,0,0,0,0,0,0,0};

  for (int kt = 0; kt < KT; ++kt) {
    v8i afrag;
    const s8* arow = act + (long long)row * K + kt * 64;
#pragma unroll
    for (int v = 0; v < 8; ++v) {
      const int koff = ((v >> 1) << 4) + (khalf << 3) + ((v & 1) << 2);
      afrag[v] = *(const int*)(arow + koff);
    }
    const long long ko = (long long)kt << 8;
    v8i b0, b1, b2, b3;
#pragma unroll
    for (int v = 0; v < 8; ++v) b0[v] = bp0[ko + v];
#pragma unroll
    for (int v = 0; v < 8; ++v) b1[v] = bp1[ko + v];
#pragma unroll
    for (int v = 0; v < 8; ++v) b2[v] = bp2[ko + v];
#pragma unroll
    for (int v = 0; v < 8; ++v) b3[v] = bp3[ko + v];
    __builtin_prefetch(bp0 + ko + 256, 0, 0);  // next K-step weight stream
    acc0 = __builtin_amdgcn_wmma_i32_16x16x64_iu8(true, afrag, true, b0, acc0, false, false);
    acc1 = __builtin_amdgcn_wmma_i32_16x16x64_iu8(true, afrag, true, b1, acc1, false, false);
    acc2 = __builtin_amdgcn_wmma_i32_16x16x64_iu8(true, afrag, true, b2, acc2, false, false);
    acc3 = __builtin_amdgcn_wmma_i32_16x16x64_iu8(true, afrag, true, b3, acc3, false, false);
  }

  const float lo = relu ? 0.0f : -128.0f;
  const int n = lane & 15;
#pragma unroll
  for (int t = 0; t < 4; ++t) {
    const v8i acc = (t == 0) ? acc0 : (t == 1) ? acc1 : (t == 2) ? acc2 : acc3;
    const int oct = oct0 + t;
    if (oct >= OCT) continue;
    const int oc = oct * 16 + n;
    if (oc >= OC) continue;
    const int bi = (int)bias[oc];
#pragma unroll
    for (int r = 0; r < 8; ++r) {
      const int brow = bt * 16 + r + (khalf << 3);
      float o = floorf((float)(acc[r] + bi) * M);
      o = fminf(fmaxf(o, lo), 127.0f);
      if (out_s8) out_s8[(long long)brow * OC + oc] = (s8)(int)o;
      else        out_f32[(long long)brow * OC + oc] = o;
    }
  }
}

// ---------------------------------------------------------------------------
// 3x3 stride-2 VALID max pool on s8 NCHW
// ---------------------------------------------------------------------------
__global__ void maxpool_kernel(const s8* __restrict__ in, s8* __restrict__ out,
                               int B, int C, int H, int W, int OH, int OW) {
  long long i = (long long)blockIdx.x * 256 + threadIdx.x;
  long long total = (long long)B * C * OH * OW;
  if (i >= total) return;
  int ox = (int)(i % OW); long long t = i / OW;
  int oy = (int)(t % OH); t /= OH;
  int c  = (int)(t % C);
  int b  = (int)(t / C);
  const s8* base = in + (((long long)b * C + c) * H + oy * 2) * W + ox * 2;
  int mx = -128;
#pragma unroll
  for (int dy = 0; dy < 3; ++dy)
#pragma unroll
    for (int dx = 0; dx < 3; ++dx) {
      int v = base[dy * W + dx];
      mx = v > mx ? v : mx;
    }
  out[i] = (s8)mx;
}

// adaptive avg pool (3x3 -> 1) + floor; inputs are >=0 after ReLU
__global__ void avgpool_kernel(const s8* __restrict__ in, s8* __restrict__ out, int n) {
  int i = blockIdx.x * 256 + threadIdx.x;
  if (i >= n) return;
  const s8* p = in + (long long)i * 9;
  int s = 0;
#pragma unroll
  for (int j = 0; j < 9; ++j) s += p[j];
  out[i] = (s8)(s / 9);  // floor(mean) for non-negative sums
}

// ---------------------------------------------------------------------------
// Host-side launch
// ---------------------------------------------------------------------------
extern "C" void kernel_launch(void* const* d_in, const int* in_sizes, int n_in,
                              void* d_out, int out_size, void* d_ws, size_t ws_size,
                              hipStream_t stream) {
  (void)in_sizes; (void)n_in; (void)out_size; (void)ws_size;
  const float* x   = (const float*)d_in[0];
  const float* w1  = (const float*)d_in[1];  const float* b1  = (const float*)d_in[2];
  const float* w2  = (const float*)d_in[3];  const float* b2  = (const float*)d_in[4];
  const float* w3  = (const float*)d_in[5];  const float* b3  = (const float*)d_in[6];
  const float* w4  = (const float*)d_in[7];  const float* b4  = (const float*)d_in[8];
  const float* w5  = (const float*)d_in[9];  const float* b5  = (const float*)d_in[10];
  const float* wf1 = (const float*)d_in[11]; const float* bf1 = (const float*)d_in[12];
  const float* wf2 = (const float*)d_in[13]; const float* bf2 = (const float*)d_in[14];
  const float* wf3 = (const float*)d_in[15]; const float* bf3 = (const float*)d_in[16];
  float* out = (float*)d_out;

  char* ws = (char*)d_ws;
  size_t off = 0;
  auto take = [&](size_t bytes) {
    size_t o = off;
    off = (off + bytes + 255) & ~(size_t)255;
    return o;
  };
  const int B = 512;
  s8* qx  = (s8*)(ws + take((size_t)B * 3 * 32 * 32));
  s8* a1  = (s8*)(ws + take((size_t)B * 96 * 32 * 32));
  s8* p1  = (s8*)(ws + take((size_t)B * 96 * 15 * 15));
  s8* a2  = (s8*)(ws + take((size_t)B * 256 * 15 * 15));
  s8* p2  = (s8*)(ws + take((size_t)B * 256 * 7 * 7));
  s8* a3  = (s8*)(ws + take((size_t)B * 384 * 7 * 7));
  s8* a4  = (s8*)(ws + take((size_t)B * 384 * 7 * 7));
  s8* a5  = (s8*)(ws + take((size_t)B * 256 * 7 * 7));
  s8* p3  = (s8*)(ws + take((size_t)B * 256 * 3 * 3));
  s8* gap = (s8*)(ws + take((size_t)B * 256));
  s8* f1  = (s8*)(ws + take((size_t)B * 4096));
  s8* f2  = (s8*)(ws + take((size_t)B * 4096));

  auto frag_bytes = [](int R, int K) {
    return (size_t)((R + 15) / 16) * ((K + 63) / 64) * 1024;
  };
  s8* pw1 = (s8*)(ws + take(frag_bytes(96, 75)));
  s8* pw2 = (s8*)(ws + take(frag_bytes(256, 2400)));
  s8* pw3 = (s8*)(ws + take(frag_bytes(384, 2304)));
  s8* pw4 = (s8*)(ws + take(frag_bytes(384, 3456)));
  s8* pw5 = (s8*)(ws + take(frag_bytes(256, 3456)));
  s8* pf1 = (s8*)(ws + take(frag_bytes(4096, 256)));
  s8* pf2 = (s8*)(ws + take(frag_bytes(4096, 4096)));
  s8* pf3 = (s8*)(ws + take(frag_bytes(10, 4096)));

  // --- quantize input + pack weights ---
  {
    int n = B * 3 * 32 * 32;
    quantize_kernel<<<(n + 255) / 256, 256, 0, stream>>>(x, qx, n);
  }
  auto pack = [&](const float* w, s8* dst, int R, int K) {
    long long tot = (long long)frag_bytes(R, K);
    pack_frag_kernel<<<(unsigned)((tot + 255) / 256), 256, 0, stream>>>(w, dst, R, K);
  };
  pack(w1, pw1, 96, 75);
  pack(w2, pw2, 256, 2400);
  pack(w3, pw3, 384, 2304);
  pack(w4, pw4, 384, 3456);
  pack(w5, pw5, 256, 3456);
  pack(wf1, pf1, 4096, 256);
  pack(wf2, pf2, 4096, 4096);
  pack(wf3, pf3, 10, 4096);

  // --- conv1: [B,3,32,32] -> [B,96,32,32]  (OCT=6 -> grid.y=1) ---
  {
    int npix = B * 32 * 32;
    conv_wmma_kernel<5, 5, 2><<<dim3(npix / 16, 1), 128, 0, stream>>>(
        qx, pw1, b1, a1, B, 3, 32, 32, 96, 2e-3f);
  }
  {
    long long tot = (long long)B * 96 * 15 * 15;
    maxpool_kernel<<<(unsigned)((tot + 255) / 256), 256, 0, stream>>>(a1, p1, B, 96, 32, 32, 15, 15);
  }
  // --- conv2: [B,96,15,15] -> [B,256,15,15]  (OCT=16 -> grid.y=1) ---
  {
    int npix = B * 15 * 15;
    conv_wmma_kernel<5, 5, 2><<<dim3(npix / 16, 1), 128, 0, stream>>>(
        p1, pw2, b2, a2, B, 96, 15, 15, 256, 4e-4f);
  }
  {
    long long tot = (long long)B * 256 * 7 * 7;
    maxpool_kernel<<<(unsigned)((tot + 255) / 256), 256, 0, stream>>>(a2, p2, B, 256, 15, 15, 7, 7);
  }
  // --- conv3..5: 3x3 pad 1 on 7x7 ---
  {
    int npix = B * 7 * 7;
    conv_wmma_kernel<3, 3, 1><<<dim3(npix / 16, 2), 128, 0, stream>>>(
        p2, pw3, b3, a3, B, 256, 7, 7, 384, 2e-4f);   // OCT=24
    conv_wmma_kernel<3, 3, 1><<<dim3(npix / 16, 2), 128, 0, stream>>>(
        a3, pw4, b4, a4, B, 384, 7, 7, 384, 2e-4f);   // OCT=24
    conv_wmma_kernel<3, 3, 1><<<dim3(npix / 16, 1), 128, 0, stream>>>(
        a4, pw5, b5, a5, B, 384, 7, 7, 256, 2e-4f);   // OCT=16
  }
  {
    long long tot = (long long)B * 256 * 3 * 3;
    maxpool_kernel<<<(unsigned)((tot + 255) / 256), 256, 0, stream>>>(a5, p3, B, 256, 7, 7, 3, 3);
  }
  {
    int n = B * 256;
    avgpool_kernel<<<(n + 255) / 256, 256, 0, stream>>>(p3, gap, n);
  }
  // --- FC layers ---
  fc_wmma_kernel<<<dim3(B / 16, 16), 128, 0, stream>>>(gap, pf1, bf1, f1, nullptr,
                                                       256, 4096, 1e-3f, 1);
  fc_wmma_kernel<<<dim3(B / 16, 16), 128, 0, stream>>>(f1, pf2, bf2, f2, nullptr,
                                                       4096, 4096, 2e-4f, 1);
  fc_wmma_kernel<<<dim3(B / 16, 1), 128, 0, stream>>>(f2, pf3, bf3, nullptr, out,
                                                      4096, 10, 2e-4f, 0);
}
